// CorrelatedAttentionBlock_81286551044296
// MI455X (gfx1250) — compile-verified, matrix operand order checked by reading, not run on hardware
//
#include <hip/hip_runtime.h>

// ---------------------------------------------------------------------------
// CorrelatedAttentionBlock for MI455X (gfx1250, wave32, WMMA bf16 -> f32)
// B=4, C=256, T=512, F=64, H=8, Dh=32
// Round 2: + async global->LDS staging (ASYNCcnt path) for pure-bf16 copies,
//          + global_prefetch of next k-step x tile in the QKV GEMM.
// ---------------------------------------------------------------------------

#define DEVINL __device__ __forceinline__

typedef __attribute__((ext_vector_type(8)))  __bf16 bf16x8;
typedef __attribute__((ext_vector_type(16))) __bf16 bf16x16;
typedef __attribute__((ext_vector_type(8)))  float  f32x8;

constexpr int B_  = 4;
constexpr int C_  = 256;
constexpr int T_  = 512;
constexpr int F_  = 64;
constexpr int H_  = 8;
constexpr int Dh_ = 32;
constexpr int RPB  = T_ * F_;      // rows per batch in the (b,t,f) flattening = 32768
constexpr int ROWS = B_ * RPB;     // 131072 total GEMM rows

DEVINL f32x8 wmma_bf16(bf16x16 a, bf16x16 b, f32x8 c) {
  // D = A(16x32) * B(32x16) + C, f32 accumulate
  return __builtin_amdgcn_wmma_f32_16x16x32_bf16(
      /*neg_a=*/false, a, /*neg_b=*/false, b,
      /*c_mod=*/(short)0, c, /*reuse_a=*/false, /*reuse_b=*/false);
}

// Row-major fragment load from LDS: operand element (r, k) stored at
// base[r*ld + k], r in [0,16), k in [0,32). ld must be a multiple of 8
// elements and base 16B-aligned so each 8-element k-group is one b128 load.
// Layout per CDNA5 ISA 7.12.2 (16-bit A/B): lane l -> r = l&15,
// elems 0..7 : k = 8*(l>>4) + i ; elems 8..15 : k = 16 + 8*(l>>4) + (i-8).
DEVINL bf16x16 frag_ld_rowmajor(const __bf16* base, int ld) {
  const int l  = threadIdx.x & 31;
  const __bf16* p = base + (l & 15) * ld + ((l >> 4) << 3);
  bf16x8 lo = *(const bf16x8*)(p);
  bf16x8 hi = *(const bf16x8*)(p + 16);
  return __builtin_shufflevector(lo, hi, 0,1,2,3,4,5,6,7,8,9,10,11,12,13,14,15);
}

// Transposed-gather fragment: logical operand element (r, k) stored at
// base[k*ld + col0 + r]  (i.e. memory is [k][r]).
DEVINL bf16x16 frag_ld_gatherT(const __bf16* base, int ld, int col0) {
  const int l  = threadIdx.x & 31;
  const int r  = l & 15;
  const int ko = (l >> 4) << 3;
  bf16x16 v;
#pragma unroll
  for (int i = 0; i < 8; ++i) {
    v[i]     = base[(ko + i) * ld + col0 + r];
    v[i + 8] = base[(16 + ko + i) * ld + col0 + r];
  }
  return v;
}

DEVINL f32x8 f32x8_zero() {
  f32x8 z = {0.f, 0.f, 0.f, 0.f, 0.f, 0.f, 0.f, 0.f};
  return z;
}

// Async DMA 16 bytes global -> LDS (raw copy, ASYNCcnt-tracked).
// lds_off: byte offset within the wave's LDS allocation (low 32 bits of the
// generic pointer to __shared__), gp: global source address.
DEVINL void async_copy_b128(unsigned lds_off, const void* gp) {
  asm volatile("global_load_async_to_lds_b128 %0, %1, off"
               :: "v"(lds_off), "v"(gp)
               : "memory");
}

DEVINL void async_wait_all() {
  asm volatile("s_wait_asynccnt 0x0" ::: "memory");
}

// ---------------------------------------------------------------------------
// Kernel 1: Q/K/V projections.  out[r, n] = sum_c xf[r,c] * W[n,c] + b[n]
// xf[(b,t,f), c] == x[b, c, t, f]  (rows contiguous in memory for fixed c).
// z-loop over the three weight sets inside the block keeps A-tile re-reads
// temporally close -> served by the 192MB L2 instead of HBM.
// Results stored bf16 in (B,H,F,T,Dh) layout for the attention kernel.
// ---------------------------------------------------------------------------
__global__ __launch_bounds__(256)
void qkv_proj_kernel(const float* __restrict__ x,
                     const float* __restrict__ Wq, const float* __restrict__ bq,
                     const float* __restrict__ Wk, const float* __restrict__ bk,
                     const float* __restrict__ Wv, const float* __restrict__ bv,
                     __bf16* __restrict__ Qws, __bf16* __restrict__ Kws,
                     __bf16* __restrict__ Vws) {
  __shared__ __align__(16) __bf16 As[128 * 40];
  __shared__ __align__(16) __bf16 Bs[128 * 40];

  const int tid  = threadIdx.x;
  const int lane = tid & 31;
  const int wave = tid >> 5;
  const int wm   = wave & 3;   // 4 waves along M (32 rows each)
  const int wn   = wave >> 2;  // 2 waves along N (64 cols each)
  const int row0 = blockIdx.x * 128;
  const int col0 = blockIdx.y * 128;
  const int b    = row0 >> 15;          // 128-row tiles never cross a batch
  const int rb   = row0 & (RPB - 1);    // flattened (t,f) base within batch

  const float* Ws[3] = {Wq, Wk, Wv};
  const float* bvs[3] = {bq, bk, bv};
  __bf16* Ds[3] = {Qws, Kws, Vws};

  for (int z = 0; z < 3; ++z) {
    const float* W    = Ws[z];
    const float* bias = bvs[z];
    __bf16* dst       = Ds[z];

    f32x8 acc[2][4];
#pragma unroll
    for (int mi = 0; mi < 2; ++mi)
#pragma unroll
      for (int ni = 0; ni < 4; ++ni) acc[mi][ni] = f32x8_zero();

    for (int k0 = 0; k0 < C_; k0 += 32) {
      // A tile: 128 rows x 32 c.  For fixed c the 128 rows are contiguous.
      for (int i = tid; i < 128 * 32; i += 256) {
        int lc = i >> 7;        // 0..31 (c within k-step)
        int lr = i & 127;       // 0..127 (row)
        float v = x[(size_t)(b * C_ + (k0 + lc)) * RPB + rb + lr];
        As[lr * 40 + lc] = (__bf16)v;
      }
      // B tile: 128 n x 32 c.  B[n][c] = W[(col0+n)*C + k0+c].
      for (int i = tid; i < 128 * 32; i += 256) {
        int ln = i >> 5;
        int lc = i & 31;
        Bs[ln * 40 + lc] = (__bf16)W[(size_t)(col0 + ln) * C_ + k0 + lc];
      }
      __syncthreads();

      // Prefetch next k-step's A tile (x is fp32, needs VALU conversion, so
      // warm the cache hierarchy while the matrix units run).
      if (k0 + 32 < C_) {
        const float* nx = &x[(size_t)(b * C_ + (k0 + 32 + (tid >> 3))) * RPB +
                             rb + ((tid & 7) << 4)];
        __builtin_prefetch(nx, 0, 1);
      }

      bf16x16 af[2], bfv[4];
#pragma unroll
      for (int mi = 0; mi < 2; ++mi)
        af[mi] = frag_ld_rowmajor(&As[(wm * 32 + mi * 16) * 40], 40);
#pragma unroll
      for (int ni = 0; ni < 4; ++ni)
        bfv[ni] = frag_ld_rowmajor(&Bs[(wn * 64 + ni * 16) * 40], 40);
#pragma unroll
      for (int mi = 0; mi < 2; ++mi)
#pragma unroll
        for (int ni = 0; ni < 4; ++ni)
          acc[mi][ni] = wmma_bf16(af[mi], bfv[ni], acc[mi][ni]);
      __syncthreads();
    }

    // Epilogue: bias + scatter into (B,H,F,T,Dh) bf16.
#pragma unroll
    for (int mi = 0; mi < 2; ++mi)
#pragma unroll
      for (int ni = 0; ni < 4; ++ni)
#pragma unroll
        for (int r = 0; r < 8; ++r) {
          int m = row0 + wm * 32 + mi * 16 + r + ((lane >> 4) << 3);
          int n = col0 + wn * 64 + ni * 16 + (lane & 15);
          float v = acc[mi][ni][r] + bias[n];
          int rem = m & (RPB - 1);
          int t = rem >> 6, f = rem & 63;
          int h = n >> 5, d = n & 31;
          size_t o = ((((size_t)(m >> 15) * H_ + h) * F_ + f) * T_ + t) * Dh_ + d;
          dst[o] = (__bf16)v;
        }
    __syncthreads();  // LDS reused by next z
  }
}

// ---------------------------------------------------------------------------
// Kernel 2: per-(b,h,f) correlated attention.  One block per (b,h,f).
//   1) column sum-of-squares over T for K and Q  -> inv norms
//   2) cov[d,e] = sum_t Khat[t,d] * Qhat[t,e]  (WMMA, 8 waves split t)
//   3) softmax over e with tau = clip(exp(log_tau), 1e-4, 10)
//   4) out[t,e] = sum_d V[t,d] * att[d,e]      (WMMA, V staged via async DMA)
// Output written bf16 in (B,T,F,C) layout for the final projection.
// ---------------------------------------------------------------------------
__global__ __launch_bounds__(256)
void corr_attn_kernel(const __bf16* __restrict__ Qws,
                      const __bf16* __restrict__ Kws,
                      const __bf16* __restrict__ Vws,
                      const float* __restrict__ log_tau,
                      __bf16* __restrict__ outws) {
  // Phase-overlapped LDS layout (bytes):
  //   [0,20480)      Ks[256][40] bf16   (phase A)   / covp[8][1024] f32 / Vs (phase B)
  //   [20480,40960)  Qs[256][40] bf16   (phase A)
  //   [40960,41216)  norm2[2][32] f32
  //   [41216,41472)  inv[2][32]   f32
  //   [41472,45568)  cov[32][32]  f32
  //   [45568,48128)  attT[32][40] bf16  (att transposed: attT[e][d])
  __shared__ __align__(16) unsigned char smem[48128];
  __bf16* Ks    = (__bf16*)(smem);
  __bf16* Qs    = (__bf16*)(smem + 20480);
  float*  covp  = (float*)(smem);        // reused after phase A
  __bf16* Vs    = (__bf16*)(smem);       // reused in phase B
  float*  norm2 = (float*)(smem + 40960);
  float*  invn  = (float*)(smem + 41216);
  float*  cov   = (float*)(smem + 41472);
  __bf16* attT  = (__bf16*)(smem + 45568);

  const int tid  = threadIdx.x;
  const int lane = tid & 31;
  const int wave = tid >> 5;

  const int bhf = blockIdx.x;            // = (b*H + h)*F + f
  const int f = bhf & 63;
  const int h = (bhf >> 6) & 7;
  const int b = bhf >> 9;

  const __bf16* Qg = Qws + (size_t)bhf * (T_ * Dh_);
  const __bf16* Kg = Kws + (size_t)bhf * (T_ * Dh_);
  const __bf16* Vg = Vws + (size_t)bhf * (T_ * Dh_);

  // ---- 1) column norms over T ----
  if (tid < 64) norm2[tid] = 0.f;
  __syncthreads();
  {
    int col = tid & 31;
    int grp = (tid >> 5) & 3;        // 4 row-groups of 128
    int mat = tid >> 7;              // 0: K, 1: Q
    const __bf16* src = mat ? Qg : Kg;
    float s = 0.f;
    for (int r = grp * 128; r < grp * 128 + 128; ++r) {
      float v = (float)src[r * Dh_ + col];
      s += v * v;
    }
    atomicAdd(&norm2[mat * 32 + col], s);
  }
  __syncthreads();
  if (tid < 64) invn[tid] = rsqrtf(fmaxf(norm2[tid], 1e-8f));
  __syncthreads();

  // ---- 2) cov accumulation over 2 chunks of 256 t-values ----
  f32x8 cacc[2][2];
#pragma unroll
  for (int i = 0; i < 2; ++i)
#pragma unroll
    for (int j = 0; j < 2; ++j) cacc[i][j] = f32x8_zero();

  for (int chunk = 0; chunk < 2; ++chunk) {
    int t0 = chunk * 256;
    // stage normalized K,Q chunks as bf16 (needs scaling -> VALU path)
    for (int i = tid; i < 256 * 32; i += 256) {
      int r = i >> 5, c = i & 31;
      Ks[r * 40 + c] = (__bf16)((float)Kg[(t0 + r) * Dh_ + c] * invn[c]);
      Qs[r * 40 + c] = (__bf16)((float)Qg[(t0 + r) * Dh_ + c] * invn[32 + c]);
    }
    __syncthreads();

    // each wave owns one 32-wide t-slice == one WMMA k-step
    int toff = wave * 32;
    bf16x16 af[2], bfr[2];
#pragma unroll
    for (int dt = 0; dt < 2; ++dt)
      af[dt] = frag_ld_gatherT(Ks + toff * 40, 40, dt * 16);   // A = Khat^T
#pragma unroll
    for (int et = 0; et < 2; ++et)
      bfr[et] = frag_ld_gatherT(Qs + toff * 40, 40, et * 16);  // B = Qhat
#pragma unroll
    for (int dt = 0; dt < 2; ++dt)
#pragma unroll
      for (int et = 0; et < 2; ++et)
        cacc[dt][et] = wmma_bf16(af[dt], bfr[et], cacc[dt][et]);
    __syncthreads();
  }

  // cross-wave reduction of partial covariances
#pragma unroll
  for (int dt = 0; dt < 2; ++dt)
#pragma unroll
    for (int et = 0; et < 2; ++et)
#pragma unroll
      for (int r = 0; r < 8; ++r) {
        int d = dt * 16 + r + ((lane >> 4) << 3);
        int e = et * 16 + (lane & 15);
        covp[wave * 1024 + d * 32 + e] = cacc[dt][et][r];
      }
  __syncthreads();
  for (int i = tid; i < 1024; i += 256) {
    float s = 0.f;
#pragma unroll
    for (int w = 0; w < 8; ++w) s += covp[w * 1024 + i];
    cov[i] = s;
  }
  __syncthreads();

  // ---- 3) softmax over e, scaled by 1/tau; store transposed as bf16 ----
  if (tid < 32) {
    float tau  = fminf(fmaxf(__expf(log_tau[0]), 1e-4f), 10.0f);
    float it   = 1.0f / tau;
    float rowv[32];
    float mx = -3.4e38f;
#pragma unroll
    for (int e = 0; e < 32; ++e) {
      rowv[e] = cov[tid * 32 + e] * it;
      mx = fmaxf(mx, rowv[e]);
    }
    float sum = 0.f;
#pragma unroll
    for (int e = 0; e < 32; ++e) {
      rowv[e] = __expf(rowv[e] - mx);
      sum += rowv[e];
    }
    float isum = 1.0f / sum;
#pragma unroll
    for (int e = 0; e < 32; ++e) attT[e * 40 + tid] = (__bf16)(rowv[e] * isum);
  }
  __syncthreads();

  // ---- 4) out = V @ att; V chunk staged by async global->LDS DMA ----
  const unsigned vs_base = (unsigned)(size_t)(void*)Vs;  // LDS byte offset
  for (int chunk = 0; chunk < 2; ++chunk) {
    int t0 = chunk * 256;
    // 256 rows x 64B, row-padded to 80B in LDS: 4 x b128 per row.
    for (int i = tid; i < 1024; i += 256) {
      int r = i >> 2, q = i & 3;
      const __bf16* gp = Vg + (size_t)(t0 + r) * Dh_ + q * 8;
      async_copy_b128(vs_base + (unsigned)(r * 80 + q * 16), gp);
    }
    async_wait_all();
    __syncthreads();

    int toff = wave * 32;   // 32 output rows (t) per wave
    bf16x16 vf[2], atf[2];
#pragma unroll
    for (int mi = 0; mi < 2; ++mi)
      vf[mi] = frag_ld_rowmajor(Vs + (toff + mi * 16) * 40, 40);
#pragma unroll
    for (int ni = 0; ni < 2; ++ni)
      atf[ni] = frag_ld_rowmajor(attT + ni * 16 * 40, 40);  // B[k=d][n=e] from attT[e][d]

    f32x8 oacc[2][2];
#pragma unroll
    for (int i = 0; i < 2; ++i)
#pragma unroll
      for (int j = 0; j < 2; ++j) oacc[i][j] = f32x8_zero();
#pragma unroll
    for (int mi = 0; mi < 2; ++mi)
#pragma unroll
      for (int ni = 0; ni < 2; ++ni)
        oacc[mi][ni] = wmma_bf16(vf[mi], atf[ni], oacc[mi][ni]);

    // store bf16 to (B,T,F,C); channel = h*Dh + e
#pragma unroll
    for (int mi = 0; mi < 2; ++mi)
#pragma unroll
      for (int ni = 0; ni < 2; ++ni)
#pragma unroll
        for (int r = 0; r < 8; ++r) {
          int t = t0 + toff + mi * 16 + r + ((lane >> 4) << 3);
          int e = ni * 16 + (lane & 15);
          size_t row = ((size_t)b * T_ + t) * F_ + f;
          outws[row * C_ + h * Dh_ + e] = (__bf16)oacc[mi][ni][r];
        }
    __syncthreads();
  }
}

// ---------------------------------------------------------------------------
// Kernel 3: output projection + transpose store.
//   out[b, n, t, f] = sum_c A[(b,t,f), c] * Wo[n, c] + bo[n]   (fp32 output)
// A is already bf16 in workspace -> staged by async global->LDS DMA.
// ---------------------------------------------------------------------------
__global__ __launch_bounds__(256)
void out_proj_kernel(const __bf16* __restrict__ Aws,
                     const float* __restrict__ Wo,
                     const float* __restrict__ bo,
                     float* __restrict__ out) {
  __shared__ __align__(16) __bf16 As[128 * 40];
  __shared__ __align__(16) __bf16 Bs[128 * 40];

  const int tid  = threadIdx.x;
  const int lane = tid & 31;
  const int wave = tid >> 5;
  const int wm   = wave & 3;
  const int wn   = wave >> 2;
  const int row0 = blockIdx.x * 128;
  const int col0 = blockIdx.y * 128;
  const unsigned as_base = (unsigned)(size_t)(void*)As;  // LDS byte offset

  f32x8 acc[2][4];
#pragma unroll
  for (int mi = 0; mi < 2; ++mi)
#pragma unroll
    for (int ni = 0; ni < 4; ++ni) acc[mi][ni] = f32x8_zero();

  for (int k0 = 0; k0 < C_; k0 += 32) {
    // A tile: async DMA 128 rows x 64B (bf16), padded to 80B rows in LDS.
    for (int i = tid; i < 512; i += 256) {
      int lr = i >> 2, q = i & 3;
      const __bf16* gp = Aws + (size_t)(row0 + lr) * C_ + k0 + q * 8;
      async_copy_b128(as_base + (unsigned)(lr * 80 + q * 16), gp);
    }
    // B tile: weights need fp32 -> bf16 conversion (VALU path).
    for (int i = tid; i < 128 * 32; i += 256) {
      int ln = i >> 5;
      int lc = i & 31;
      Bs[ln * 40 + lc] = (__bf16)Wo[(size_t)(col0 + ln) * C_ + k0 + lc];
    }
    async_wait_all();
    __syncthreads();

    bf16x16 af[2], bfv[4];
#pragma unroll
    for (int mi = 0; mi < 2; ++mi)
      af[mi] = frag_ld_rowmajor(&As[(wm * 32 + mi * 16) * 40], 40);
#pragma unroll
    for (int ni = 0; ni < 4; ++ni)
      bfv[ni] = frag_ld_rowmajor(&Bs[(wn * 64 + ni * 16) * 40], 40);
#pragma unroll
    for (int mi = 0; mi < 2; ++mi)
#pragma unroll
      for (int ni = 0; ni < 4; ++ni)
        acc[mi][ni] = wmma_bf16(af[mi], bfv[ni], acc[mi][ni]);
    __syncthreads();
  }

  // Epilogue: bias + transpose store to (B,C,T,F) fp32.
#pragma unroll
  for (int mi = 0; mi < 2; ++mi)
#pragma unroll
    for (int ni = 0; ni < 4; ++ni)
#pragma unroll
      for (int r = 0; r < 8; ++r) {
        int m = row0 + wm * 32 + mi * 16 + r + ((lane >> 4) << 3);
        int n = col0 + wn * 64 + ni * 16 + (lane & 15);
        float v = acc[mi][ni][r] + bo[n];
        int rem = m & (RPB - 1);
        int t = rem >> 6, fq = rem & 63;
        size_t o = (((size_t)(m >> 15) * C_ + n) * T_ + t) * F_ + fq;
        out[o] = v;
      }
}

// ---------------------------------------------------------------------------
// Launch
// ---------------------------------------------------------------------------
extern "C" void kernel_launch(void* const* d_in, const int* in_sizes, int n_in,
                              void* d_out, int out_size, void* d_ws,
                              size_t ws_size, hipStream_t stream) {
  const float* x       = (const float*)d_in[0];
  const float* Wq      = (const float*)d_in[1];
  const float* bq      = (const float*)d_in[2];
  const float* Wk      = (const float*)d_in[3];
  const float* bk      = (const float*)d_in[4];
  const float* Wv      = (const float*)d_in[5];
  const float* bv      = (const float*)d_in[6];
  const float* Wo      = (const float*)d_in[7];
  const float* bo      = (const float*)d_in[8];
  const float* log_tau = (const float*)d_in[9];

  // Workspace: Q | K | V in (B,H,F,T,Dh) bf16, then mixed output in (B,T,F,C) bf16.
  const size_t NQ = (size_t)B_ * H_ * F_ * T_ * Dh_;  // 33,554,432 elems (64 MB bf16)
  __bf16* Qws = (__bf16*)d_ws;
  __bf16* Kws = Qws + NQ;
  __bf16* Vws = Kws + NQ;
  __bf16* Ows = Vws + NQ;

  dim3 blk(256, 1, 1);
  qkv_proj_kernel<<<dim3(ROWS / 128, C_ / 128, 1), blk, 0, stream>>>(
      x, Wq, bq, Wk, bk, Wv, bv, Qws, Kws, Vws);
  corr_attn_kernel<<<dim3(B_ * H_ * F_, 1, 1), blk, 0, stream>>>(
      Qws, Kws, Vws, log_tau, Ows);
  out_proj_kernel<<<dim3(ROWS / 128, C_ / 128, 1), blk, 0, stream>>>(
      Ows, Wo, bo, (float*)d_out);
}
